// MultiHeadAttention_47717086658582
// MI455X (gfx1250) — matrix operand order repaired and checked by reading, not compiled, not run on hardware
//
#include <hip/hip_runtime.h>

// ---------------- problem constants ----------------
#define BN      2
#define SEQ     2048
#define DMODEL  1024
#define NHEADS  16
#define KDIM    64
#define MTOT    (BN * SEQ)        // 4096 rows

// ---------------- types ----------------
typedef __attribute__((ext_vector_type(16))) __bf16 v16bf;
typedef __attribute__((ext_vector_type(8)))  float  v8f;
typedef __attribute__((ext_vector_type(4)))  unsigned int u32x4;
typedef __attribute__((ext_vector_type(8)))  int    i32x8;
typedef __attribute__((ext_vector_type(4)))  int    i32x4;

union FragB16 {
  v16bf v;
  uint4 q[2];          // two 16-byte halves (8 bf16 each)
};

__device__ __forceinline__ v8f wmma_bf16f32(v16bf a, v16bf b, v8f c) {
  return __builtin_amdgcn_wmma_f32_16x16x32_bf16(false, a, false, b, (short)0, c, false, false);
}

__device__ __forceinline__ unsigned short f2bf(float f) {
  unsigned int u = __float_as_uint(f);
  return (unsigned short)((u + 0x7FFFu + ((u >> 16) & 1u)) >> 16);  // RNE
}

__device__ __forceinline__ v8f vzero8() {
  v8f z = {0.f, 0.f, 0.f, 0.f, 0.f, 0.f, 0.f, 0.f};
  return z;
}

#if __has_builtin(__builtin_amdgcn_tensor_load_to_lds) && __has_builtin(__builtin_amdgcn_s_wait_tensorcnt)
#define HAVE_TDM 1
#else
#define HAVE_TDM 0
#endif

#if HAVE_TDM
// LDS byte offset of a __shared__ object (generic -> addrspace(3) -> int)
__device__ __forceinline__ unsigned lds_off(const void* p) {
  return (unsigned)(unsigned long long)(__attribute__((address_space(3))) const char*)p;
}

// 2-D bf16 tile DMA: tile_k elements per line, `rows` lines, line stride in elements.
// D# per CDNA5 ISA 8.3/8.4: group0 = {count|flags, lds_addr, gaddr lo, gaddr hi | type=2},
// group1 packs data_size=1(2B), tensor_dim0/1, tile_dim0/1, tensor_dim0_stride.
// Groups 2/3 (dims >= 3) are zero for 2-D tiles; this toolchain's 6-arg builtin
// takes an extra i32x8 group (zero here) ahead of the cache-policy immediate.
__device__ __forceinline__ void tdm_load_2d(unsigned lds_byte_off, const void* gaddr,
                                            unsigned tile_k, unsigned rows,
                                            unsigned stride_elems) {
  unsigned long long ga = (unsigned long long)gaddr;
  const unsigned td0 = stride_elems;       // line length for OOB (never exceeded)
  const unsigned td1 = 0x100000u;          // large: tiles never go OOB in dim1
  u32x4 g0;
  g0[0] = 1u;                                   // count=1, user mode, no gather
  g0[1] = lds_byte_off;
  g0[2] = (unsigned)ga;
  g0[3] = (unsigned)(ga >> 32) | 0x80000000u;   // addr[56:32] | type=2 in [31:30]
  i32x8 g1;
  g1[0] = (int)(1u << 16);                      // wg_mask=0, data_size=1 (2 bytes)
  g1[1] = (int)((td0 & 0xFFFFu) << 16);         // tensor_dim0[15:0]
  g1[2] = (int)((td0 >> 16) | ((td1 & 0xFFFFu) << 16));
  g1[3] = (int)((td1 >> 16) | (tile_k << 16));  // tile_dim0
  g1[4] = (int)(rows & 0xFFFFu);                // tile_dim1; tile_dim2=0
  g1[5] = (int)stride_elems;                    // tensor_dim0_stride[31:0]
  g1[6] = 0;
  g1[7] = 0;
  i32x4 z4 = {0, 0, 0, 0};
  i32x8 z8 = {0, 0, 0, 0, 0, 0, 0, 0};
  __builtin_amdgcn_tensor_load_to_lds(g0, g1, z4, z4, z8, 0);
}
#endif

// ---------------- kernel 1: x fp32 -> bf16 ----------------
__global__ void __launch_bounds__(256)
k_convert_bf16(const float* __restrict__ in, unsigned short* __restrict__ out, int n4) {
  int i = blockIdx.x * blockDim.x + threadIdx.x;
  if (i >= n4) return;
  float4 f = ((const float4*)in)[i];
  ushort4 o;
  o.x = f2bf(f.x); o.y = f2bf(f.y); o.z = f2bf(f.z); o.w = f2bf(f.w);
  ((ushort4*)out)[i] = o;
}

// ---------------- kernel 2: W [K][N] fp32 -> Wt [N][K] bf16 ----------------
__global__ void __launch_bounds__(256)
k_transpose_bf16(const float* __restrict__ W, unsigned short* __restrict__ Wt) {
  __shared__ float tile[32][33];
  const int tx = threadIdx.x;
  const int x  = blockIdx.x * 32 + tx;
  const int y0 = blockIdx.y * 32;
  #pragma unroll
  for (int i = threadIdx.y; i < 32; i += 8)
    tile[i][tx] = W[(size_t)(y0 + i) * DMODEL + x];
  __syncthreads();
  const int xo  = blockIdx.y * 32 + tx;
  const int yo0 = blockIdx.x * 32;
  #pragma unroll
  for (int i = threadIdx.y; i < 32; i += 8)
    Wt[(size_t)(yo0 + i) * DMODEL + xo] = f2bf(tile[tx][i]);
}

// ---------------- shared GEMM core: 64x128 block tile, TDM-staged, double buffered ----------------
// Abuf: [2][64*32] halves (4 KB each), Bbuf: [2][128*32] halves (8 KB each)
__device__ __forceinline__ void gemm_core(const unsigned short* __restrict__ A,
                                          const unsigned short* __restrict__ Wt,
                                          unsigned short* Abuf, unsigned short* Bbuf,
                                          int m0, int n0, int wid, int tid,
                                          int lo, int hi, int wrow, int wcol,
                                          v8f acc[2][4]) {
#if HAVE_TDM
  if (wid == 0) {
    tdm_load_2d(lds_off(Abuf), A + (size_t)m0 * DMODEL, 32, 64, DMODEL);
    tdm_load_2d(lds_off(Bbuf), Wt + (size_t)n0 * DMODEL, 32, 128, DMODEL);
  }
  int buf = 0;
  for (int kc = 0; kc < DMODEL; kc += 32, buf ^= 1) {
    const bool has_next = (kc + 32 < DMODEL);
    if (wid == 0 && has_next) {
      tdm_load_2d(lds_off(Abuf + (buf ^ 1) * 64 * 32),
                  A + (size_t)m0 * DMODEL + kc + 32, 32, 64, DMODEL);
      tdm_load_2d(lds_off(Bbuf + (buf ^ 1) * 128 * 32),
                  Wt + (size_t)n0 * DMODEL + kc + 32, 32, 128, DMODEL);
    }
    if (has_next) __builtin_amdgcn_s_wait_tensorcnt((short)2);  // current pair retired
    else          __builtin_amdgcn_s_wait_tensorcnt((short)0);
    __syncthreads();

    const unsigned short* ab = Abuf + buf * 64 * 32;
    const unsigned short* bb = Bbuf + buf * 128 * 32;
    FragB16 a0, a1;
    const unsigned short* ap = ab + (wrow + lo) * 32 + hi * 8;
    a0.q[0] = *(const uint4*)(ap);        a0.q[1] = *(const uint4*)(ap + 16);
    a1.q[0] = *(const uint4*)(ap + 512);  a1.q[1] = *(const uint4*)(ap + 512 + 16);
    #pragma unroll
    for (int i = 0; i < 4; ++i) {
      FragB16 b;
      const unsigned short* bp = bb + (wcol + i * 16 + lo) * 32 + hi * 8;
      b.q[0] = *(const uint4*)(bp);  b.q[1] = *(const uint4*)(bp + 16);
      acc[0][i] = wmma_bf16f32(a0.v, b.v, acc[0][i]);
      acc[1][i] = wmma_bf16f32(a1.v, b.v, acc[1][i]);
    }
    __syncthreads();
  }
#else
  // Fallback: cooperative register-staged copy into single LDS buffer, prefetching
  // next slice's registers while WMMAs consume the current slice.
  uint4 ra0, ra1, rb0, rb1, rb2, rb3;
  {
    const unsigned short* asrc = A + (size_t)(m0 + (tid >> 1)) * DMODEL + (tid & 1) * 16;
    ra0 = *(const uint4*)(asrc);  ra1 = *(const uint4*)(asrc + 8);
    const unsigned short* bsrc = Wt + (size_t)(n0 + tid) * DMODEL;
    rb0 = *(const uint4*)(bsrc);      rb1 = *(const uint4*)(bsrc + 8);
    rb2 = *(const uint4*)(bsrc + 16); rb3 = *(const uint4*)(bsrc + 24);
  }
  for (int kc = 0; kc < DMODEL; kc += 32) {
    *(uint4*)(Abuf + tid * 16)     = ra0;
    *(uint4*)(Abuf + tid * 16 + 8) = ra1;
    unsigned short* bd = Bbuf + tid * 32;
    *(uint4*)(bd) = rb0; *(uint4*)(bd + 8) = rb1;
    *(uint4*)(bd + 16) = rb2; *(uint4*)(bd + 24) = rb3;
    __syncthreads();
    if (kc + 32 < DMODEL) {
      const unsigned short* asrc = A + (size_t)(m0 + (tid >> 1)) * DMODEL + kc + 32 + (tid & 1) * 16;
      ra0 = *(const uint4*)(asrc);  ra1 = *(const uint4*)(asrc + 8);
      const unsigned short* bsrc = Wt + (size_t)(n0 + tid) * DMODEL + kc + 32;
      rb0 = *(const uint4*)(bsrc);      rb1 = *(const uint4*)(bsrc + 8);
      rb2 = *(const uint4*)(bsrc + 16); rb3 = *(const uint4*)(bsrc + 24);
    }
    FragB16 a0, a1;
    const unsigned short* ap = Abuf + (wrow + lo) * 32 + hi * 8;
    a0.q[0] = *(const uint4*)(ap);        a0.q[1] = *(const uint4*)(ap + 16);
    a1.q[0] = *(const uint4*)(ap + 512);  a1.q[1] = *(const uint4*)(ap + 512 + 16);
    #pragma unroll
    for (int i = 0; i < 4; ++i) {
      FragB16 b;
      const unsigned short* bp = Bbuf + (wcol + i * 16 + lo) * 32 + hi * 8;
      b.q[0] = *(const uint4*)(bp);  b.q[1] = *(const uint4*)(bp + 16);
      acc[0][i] = wmma_bf16f32(a0.v, b.v, acc[0][i]);
      acc[1][i] = wmma_bf16f32(a1.v, b.v, acc[1][i]);
    }
    __syncthreads();
  }
#endif
}

// ---------------- kernel 3: QKV projection GEMM ----------------
__global__ void __launch_bounds__(128)
k_gemm_qkv(const unsigned short* __restrict__ A,
           const unsigned short* __restrict__ Wt,
           unsigned short* __restrict__ out, int vt_mode) {
  __shared__ __align__(16) unsigned short Abuf[2 * 64 * 32];
  __shared__ __align__(16) unsigned short Bbuf[2 * 128 * 32];
  const int tid  = threadIdx.x;
  const int lane = tid & 31;
  const int wid  = tid >> 5;
  const int lo = lane & 15, hi = lane >> 4;
  const int m0 = blockIdx.x * 64;
  const int n0 = blockIdx.y * 128;
  const int wrow = (wid & 1) * 32;
  const int wcol = (wid >> 1) * 64;

  v8f acc[2][4];
  #pragma unroll
  for (int t = 0; t < 2; ++t)
    #pragma unroll
    for (int i = 0; i < 4; ++i) acc[t][i] = vzero8();

  gemm_core(A, Wt, Abuf, Bbuf, m0, n0, wid, tid, lo, hi, wrow, wcol, acc);

  #pragma unroll
  for (int t = 0; t < 2; ++t) {
    #pragma unroll
    for (int i = 0; i < 4; ++i) {
      const int ncol = n0 + wcol + i * 16 + lo;
      const int h = ncol >> 6;
      const int d = ncol & (KDIM - 1);
      #pragma unroll
      for (int r = 0; r < 8; ++r) {
        const int m = m0 + wrow + t * 16 + r + 8 * hi;
        const int b = m >> 11;
        const int s = m & (SEQ - 1);
        const unsigned short val = f2bf(acc[t][i][r]);
        if (vt_mode) {  // V transposed: [B,H,Kd,S]
          out[(size_t)((b * NHEADS + h) * KDIM + d) * SEQ + s] = val;
        } else {        // Q/K: [B,H,S,Kd]
          out[(size_t)((b * NHEADS + h) * SEQ + s) * KDIM + d] = val;
        }
      }
    }
  }
}

// ---------------- kernel 5: out = attn(bf16) @ Wo_t -> fp32 ----------------
__global__ void __launch_bounds__(128)
k_gemm_out(const unsigned short* __restrict__ A,
           const unsigned short* __restrict__ Wt,
           float* __restrict__ out) {
  __shared__ __align__(16) unsigned short Abuf[2 * 64 * 32];
  __shared__ __align__(16) unsigned short Bbuf[2 * 128 * 32];
  const int tid  = threadIdx.x;
  const int lane = tid & 31;
  const int wid  = tid >> 5;
  const int lo = lane & 15, hi = lane >> 4;
  const int m0 = blockIdx.x * 64;
  const int n0 = blockIdx.y * 128;
  const int wrow = (wid & 1) * 32;
  const int wcol = (wid >> 1) * 64;

  v8f acc[2][4];
  #pragma unroll
  for (int t = 0; t < 2; ++t)
    #pragma unroll
    for (int i = 0; i < 4; ++i) acc[t][i] = vzero8();

  gemm_core(A, Wt, Abuf, Bbuf, m0, n0, wid, tid, lo, hi, wrow, wcol, acc);

  #pragma unroll
  for (int t = 0; t < 2; ++t) {
    #pragma unroll
    for (int i = 0; i < 4; ++i) {
      const int ncol = n0 + wcol + i * 16 + lo;
      #pragma unroll
      for (int r = 0; r < 8; ++r) {
        const int m = m0 + wrow + t * 16 + r + 8 * hi;
        out[(size_t)m * DMODEL + ncol] = acc[t][i][r];
      }
    }
  }
}

// ---------------- kernel 4: flash attention ----------------
__global__ void __launch_bounds__(128)
k_attention(const unsigned short* __restrict__ Q,
            const unsigned short* __restrict__ K,
            const unsigned short* __restrict__ Vt,
            unsigned short* __restrict__ O) {
  __shared__ __align__(16) unsigned short Plds[4][16 * 32];

  const int lane = threadIdx.x & 31;
  const int wid  = threadIdx.x >> 5;
  const int lo   = lane & 15;
  const int hi   = lane >> 4;
  const int gw   = blockIdx.x * 4 + wid;   // 0..4095
  const int bh   = gw >> 7;
  const int qt   = gw & 127;
  const int qbase = qt * 16;
  const float scale = 0.125f;

  const unsigned short* qp = Q + (size_t)(bh * SEQ + qbase + lo) * KDIM;
  FragB16 qa0, qa1;
  qa0.q[0] = *(const uint4*)(qp + hi * 16);
  qa0.q[1] = *(const uint4*)(qp + hi * 16 + 8);
  qa1.q[0] = *(const uint4*)(qp + 32 + hi * 16);
  qa1.q[1] = *(const uint4*)(qp + 32 + hi * 16 + 8);

  v8f o0 = vzero8(), o1 = vzero8(), o2 = vzero8(), o3 = vzero8();
  float mrow[8], lrow[8];
  #pragma unroll
  for (int r = 0; r < 8; ++r) { mrow[r] = -__builtin_inff(); lrow[r] = 0.f; }

  unsigned short* pl = &Plds[wid][0];
  const unsigned short* Kbh = K  + (size_t)(bh * SEQ + lo) * KDIM + hi * 16;
  const unsigned short* Vbh = Vt + (size_t)(bh * KDIM + lo) * SEQ + hi * 16;

  FragB16 kfA[4], kfB[4];

#define LOADK(DST, KC) do {                                                           \
    const unsigned short* kp = Kbh + (size_t)(KC) * KDIM;                             \
    DST[0].q[0] = *(const uint4*)(kp);                DST[0].q[1] = *(const uint4*)(kp + 8);               \
    DST[1].q[0] = *(const uint4*)(kp + 32);           DST[1].q[1] = *(const uint4*)(kp + 40);              \
    DST[2].q[0] = *(const uint4*)(kp + 16 * KDIM);    DST[2].q[1] = *(const uint4*)(kp + 16 * KDIM + 8);   \
    DST[3].q[0] = *(const uint4*)(kp + 16 * KDIM + 32); DST[3].q[1] = *(const uint4*)(kp + 16 * KDIM + 40);\
  } while (0)

#define ATTN_CHUNK(KF, KFN, KC, DONEXT) do {                                          \
    FragB16 vb0, vb1, vb2, vb3;                                                       \
    const unsigned short* vp = Vbh + (KC);                                            \
    vb0.q[0] = *(const uint4*)(vp);             vb0.q[1] = *(const uint4*)(vp + 8);             \
    vb1.q[0] = *(const uint4*)(vp + 16 * SEQ);  vb1.q[1] = *(const uint4*)(vp + 16 * SEQ + 8);  \
    vb2.q[0] = *(const uint4*)(vp + 32 * SEQ);  vb2.q[1] = *(const uint4*)(vp + 32 * SEQ + 8);  \
    vb3.q[0] = *(const uint4*)(vp + 48 * SEQ);  vb3.q[1] = *(const uint4*)(vp + 48 * SEQ + 8);  \
    v8f s0 = vzero8(), s1 = vzero8();                                                 \
    s0 = wmma_bf16f32(qa0.v, KF[0].v, s0);                                            \
    s0 = wmma_bf16f32(qa1.v, KF[1].v, s0);                                            \
    s1 = wmma_bf16f32(qa0.v, KF[2].v, s1);                                            \
    s1 = wmma_bf16f32(qa1.v, KF[3].v, s1);                                            \
    if (DONEXT) LOADK(KFN, (KC) + 32);                                                \
    _Pragma("unroll")                                                                 \
    for (int r = 0; r < 8; ++r) {                                                     \
      float v0 = s0[r] * scale, v1 = s1[r] * scale;                                   \
      float mx = fmaxf(v0, v1);                                                       \
      mx = fmaxf(mx, __shfl_xor(mx, 1, 32));                                          \
      mx = fmaxf(mx, __shfl_xor(mx, 2, 32));                                          \
      mx = fmaxf(mx, __shfl_xor(mx, 4, 32));                                          \
      mx = fmaxf(mx, __shfl_xor(mx, 8, 32));                                          \
      float mnew = fmaxf(mrow[r], mx);                                                \
      float corr = __expf(mrow[r] - mnew);                                            \
      float p0 = __expf(v0 - mnew), p1 = __expf(v1 - mnew);                           \
      float rs = p0 + p1;                                                             \
      rs += __shfl_xor(rs, 1, 32); rs += __shfl_xor(rs, 2, 32);                       \
      rs += __shfl_xor(rs, 4, 32); rs += __shfl_xor(rs, 8, 32);                       \
      lrow[r] = lrow[r] * corr + rs;                                                  \
      mrow[r] = mnew;                                                                 \
      o0[r] *= corr; o1[r] *= corr; o2[r] *= corr; o3[r] *= corr;                     \
      const int row = r + 8 * hi;                                                     \
      pl[row * 32 + lo]      = f2bf(p0);                                              \
      pl[row * 32 + 16 + lo] = f2bf(p1);                                              \
    }                                                                                 \
    asm volatile("s_wait_dscnt 0" ::: "memory");                                      \
    FragB16 pa;                                                                       \
    { const unsigned short* pp = pl + lo * 32 + hi * 16;                              \
      pa.q[0] = *(const uint4*)(pp);  pa.q[1] = *(const uint4*)(pp + 8); }            \
    o0 = wmma_bf16f32(pa.v, vb0.v, o0);                                               \
    o1 = wmma_bf16f32(pa.v, vb1.v, o1);                                               \
    o2 = wmma_bf16f32(pa.v, vb2.v, o2);                                               \
    o3 = wmma_bf16f32(pa.v, vb3.v, o3);                                               \
  } while (0)

  LOADK(kfA, 0);
  for (int kc = 0; kc < SEQ; kc += 64) {
    ATTN_CHUNK(kfA, kfB, kc, true);
    ATTN_CHUNK(kfB, kfA, kc + 32, (kc + 64 < SEQ));
  }
#undef ATTN_CHUNK
#undef LOADK

  const int b = bh >> 4;
  const int h = bh & (NHEADS - 1);
  #pragma unroll
  for (int r = 0; r < 8; ++r) {
    const float inv = 1.0f / lrow[r];
    const int s = qbase + r + 8 * hi;
    unsigned short* op = O + (size_t)(b * SEQ + s) * DMODEL + h * KDIM;
    op[ 0 + lo] = f2bf(o0[r] * inv);
    op[16 + lo] = f2bf(o1[r] * inv);
    op[32 + lo] = f2bf(o2[r] * inv);
    op[48 + lo] = f2bf(o3[r] * inv);
  }
}

// ---------------- host launcher ----------------
extern "C" void kernel_launch(void* const* d_in, const int* in_sizes, int n_in,
                              void* d_out, int out_size, void* d_ws, size_t ws_size,
                              hipStream_t stream) {
  const float* x  = (const float*)d_in[0];
  const float* Wq = (const float*)d_in[1];
  const float* Wk = (const float*)d_in[2];
  const float* Wv = (const float*)d_in[3];
  const float* Wo = (const float*)d_in[4];
  float* out = (float*)d_out;

  char* ws = (char*)d_ws;
  const size_t MB = 1024 * 1024;
  unsigned short* Xb   = (unsigned short*)(ws + 0 * MB);    // 8 MB  [4096][1024]
  unsigned short* Wqt  = (unsigned short*)(ws + 8 * MB);    // 2 MB  [1024][1024] N-major
  unsigned short* Wkt  = (unsigned short*)(ws + 10 * MB);
  unsigned short* Wvt  = (unsigned short*)(ws + 12 * MB);
  unsigned short* Wot  = (unsigned short*)(ws + 14 * MB);
  unsigned short* Qb   = (unsigned short*)(ws + 16 * MB);   // 8 MB  [B,H,S,Kd]
  unsigned short* Kb   = (unsigned short*)(ws + 24 * MB);   // 8 MB  [B,H,S,Kd]
  unsigned short* Vtb  = (unsigned short*)(ws + 32 * MB);   // 8 MB  [B,H,Kd,S]
  unsigned short* Attn = (unsigned short*)(ws + 40 * MB);   // 8 MB  [B,S,H*Kd]

  {
    int n4 = (MTOT * DMODEL) / 4;
    k_convert_bf16<<<n4 / 256, 256, 0, stream>>>(x, Xb, n4);
  }
  {
    dim3 g(32, 32), blk(32, 8);
    k_transpose_bf16<<<g, blk, 0, stream>>>(Wq, Wqt);
    k_transpose_bf16<<<g, blk, 0, stream>>>(Wk, Wkt);
    k_transpose_bf16<<<g, blk, 0, stream>>>(Wv, Wvt);
    k_transpose_bf16<<<g, blk, 0, stream>>>(Wo, Wot);
  }
  {
    dim3 g(MTOT / 64, DMODEL / 128);            // (64, 8)
    k_gemm_qkv<<<g, 128, 0, stream>>>(Xb, Wqt, Qb, 0);
    k_gemm_qkv<<<g, 128, 0, stream>>>(Xb, Wkt, Kb, 0);
    k_gemm_qkv<<<g, 128, 0, stream>>>(Xb, Wvt, Vtb, 1);
  }
  k_attention<<<1024, 128, 0, stream>>>(Qb, Kb, Vtb, Attn);
  {
    dim3 g(MTOT / 64, DMODEL / 128);
    k_gemm_out<<<g, 128, 0, stream>>>(Attn, Wot, out);
  }
}
